// SimpleTransformerBlock_10247791968529
// MI455X (gfx1250) — matrix operand order repaired and checked by reading, not compiled, not run on hardware
//
#include <hip/hip_runtime.h>
#include <hip/hip_bf16.h>

typedef _Float16 half_t;
typedef __attribute__((ext_vector_type(16))) _Float16 v16h;
typedef __attribute__((ext_vector_type(8)))  _Float16 v8h;
typedef __attribute__((ext_vector_type(8)))  float    v8f;

#define DIM   1024
#define NHEAD 16
#define HDIM  64
#define MLPD  4096
#define BB    2
#define NN    2048
#define ROWS  (BB*NN)   // 4096

// ---------------------------------------------------------------------------
// A-operand (16x32 f16, MxK) load per ISA layout:
// lane L (L<16): row M=L, halfs 0..7 = K 0..7, halfs 8..15 = K 16..23
// lane L+16    : row M=L, halfs 0..7 = K 8..15, halfs 8..15 = K 24..31
__device__ __forceinline__ v16h load_a_tile(const half_t* __restrict__ A, int lda,
                                            int row0, int k0, int lane) {
  int row = row0 + (lane & 15);
  int kb  = k0 + ((lane >> 4) << 3);
  union { v16h v; v8h h[2]; } u;
  u.h[0] = *(const v8h*)(A + (size_t)row * lda + kb);
  u.h[1] = *(const v8h*)(A + (size_t)row * lda + kb + 16);
  return u.v;
}

// B-operand (32x16 f16, KxN) load, same K order as the A operand:
// lane L (L<16): column N=L, halfs 0..7 = K 0..7, halfs 8..15 = K 16..23
// lane L+16    : column N=L, halfs 0..7 = K 8..15, halfs 8..15 = K 24..31
// W is row-major [Ntot, Ktot]; a B column == a W row -> contiguous.
__device__ __forceinline__ v16h load_b_tile(const half_t* __restrict__ W, int ldw,
                                            int n0, int k0, int lane) {
  int n  = n0 + (lane & 15);
  int kb = k0 + ((lane >> 4) << 3);
  union { v16h v; v8h h[2]; } u;
  u.h[0] = *(const v8h*)(W + (size_t)n * ldw + kb);
  u.h[1] = *(const v8h*)(W + (size_t)n * ldw + kb + 16);
  return u.v;
}

__device__ __forceinline__ v8f wmma_f16(v16h a, v16h b, v8f c) {
  return __builtin_amdgcn_wmma_f32_16x16x32_f16(false, a, false, b, (short)0, c,
                                                false, false);
}

// CDNA5 async global->LDS copy, 16B per lane, tracked by ASYNCcnt.
__device__ __forceinline__ void async_b128(unsigned int lds_addr, const half_t* g) {
  asm volatile("global_load_async_to_lds_b128 %0, %1, off"
               :: "v"(lds_addr), "v"(g) : "memory");
}

// ---------------------------------------------------------------------------
// fp32 -> f16 conversion (weights)
__global__ void __launch_bounds__(256) f32_to_f16_kernel(const float* __restrict__ in,
                                                         half_t* __restrict__ out, int n) {
  int i = blockIdx.x * 256 + threadIdx.x;
  if (i < n) out[i] = (half_t)in[i];
}

// ---------------------------------------------------------------------------
// LayerNorm over DIM=1024, f16 output. One 256-thread block per row.
__global__ void __launch_bounds__(256) ln_f16_kernel(const float* __restrict__ x,
                                                     const float* __restrict__ w,
                                                     const float* __restrict__ b,
                                                     half_t* __restrict__ out) {
  __shared__ float red[256];
  const int tid = threadIdx.x;
  const float* xr = x + (size_t)blockIdx.x * DIM;
  float s = 0.f;
  for (int i = tid; i < DIM; i += 256) s += xr[i];
  red[tid] = s; __syncthreads();
  for (int off = 128; off > 0; off >>= 1) {
    if (tid < off) red[tid] += red[tid + off];
    __syncthreads();
  }
  const float mu = red[0] * (1.0f / DIM);
  __syncthreads();
  float v = 0.f;
  for (int i = tid; i < DIM; i += 256) { float d = xr[i] - mu; v += d * d; }
  red[tid] = v; __syncthreads();
  for (int off = 128; off > 0; off >>= 1) {
    if (tid < off) red[tid] += red[tid + off];
    __syncthreads();
  }
  const float inv = rsqrtf(red[0] * (1.0f / DIM) + 1e-5f);
  half_t* orow = out + (size_t)blockIdx.x * DIM;
  for (int i = tid; i < DIM; i += 256)
    orow[i] = (half_t)((xr[i] - mu) * inv * w[i] + b[i]);
}

// ---------------------------------------------------------------------------
// WMMA GEMM: C[M,N] = A[M,K](f16) @ W[N,K](f16)^T.
// Block tile 128x64 (4 waves x 32x64). B-chunks (64 rows x 32 K) are staged
// in LDS by async global->LDS copies, double-buffered, shared by all 4 waves.
// EPI 0: QKV scatter (q scaled by 1/8; v stored transposed [hd][n])
// EPI 1: outf = resid + C + bias       (fp32)
// EPI 2: outh = silu(C + bias)         (f16)
// EPI 3: outf = resid + C + bias       (fp32, final)
template <int EPI>
__global__ void __launch_bounds__(128) gemm_wmma_kernel(
    const half_t* __restrict__ A, const half_t* __restrict__ W,
    const float* __restrict__ bias, const float* __restrict__ resid,
    float* __restrict__ outf, half_t* __restrict__ outh,
    half_t* __restrict__ qh, half_t* __restrict__ kh, half_t* __restrict__ vt,
    int K, int N) {
  __shared__ half_t bsh[2][64 * 32];   // 2 x 4KB double buffer
  const int lane = threadIdx.x & 31;
  const int wave = threadIdx.x >> 5;
  const int tid  = threadIdx.x;
  const int m0 = blockIdx.y * 128 + wave * 32;
  const int n0 = blockIdx.x * 64;

  v8f zero = {};
  v8f acc[2][4] = {{zero, zero, zero, zero}, {zero, zero, zero, zero}};

  // stage W[n0..n0+63][k0..k0+31] into bsh[buf]; 256 16B segments, 2 per thread
  auto issue_copy = [&](int buf, int k0) {
    unsigned int base = (unsigned int)(size_t)(&bsh[buf][0]);
#pragma unroll
    for (int r = 0; r < 2; ++r) {
      int s = tid + r * 128;
      int row = s >> 2;
      int colh = (s & 3) << 3;
      async_b128(base + (unsigned int)s * 16u,
                 W + (size_t)(n0 + row) * K + k0 + colh);
    }
  };

  issue_copy(0, 0);
  const int nchunks = K >> 5;
  for (int c = 0; c < nchunks; ++c) {
    const int cur = c & 1;
    if (c + 1 < nchunks) {
      issue_copy(cur ^ 1, (c + 1) << 5);
      asm volatile("s_wait_asynccnt 2" ::: "memory");  // prev chunk landed
    } else {
      asm volatile("s_wait_asynccnt 0" ::: "memory");
    }
    __syncthreads();                                   // all waves' copies visible

    const int k0 = c << 5;
    const half_t* Bs = &bsh[cur][0];
    v16h a0 = load_a_tile(A, K, m0, k0, lane);
    v16h a1 = load_a_tile(A, K, m0 + 16, k0, lane);
#pragma unroll
    for (int t = 0; t < 4; ++t) {
      v16h b = load_b_tile(Bs, 32, t * 16, 0, lane);
      acc[0][t] = wmma_f16(a0, b, acc[0][t]);
      acc[1][t] = wmma_f16(a1, b, acc[1][t]);
    }
    __syncthreads();                                   // done reading buf before overwrite
  }

  const int nlane = lane & 15;
#pragma unroll
  for (int sub = 0; sub < 2; ++sub) {
    const int mbase = m0 + sub * 16 + 8 * (lane >> 4);
#pragma unroll
    for (int t = 0; t < 4; ++t) {
#pragma unroll
      for (int vg = 0; vg < 8; ++vg) {
        float c = acc[sub][t][vg];
        int m = mbase + vg;
        int o = n0 + t * 16 + nlane;
        if (EPI == 0) {
          int sec = o >> 10, oo = o & 1023;
          int head = oo >> 6, hd = oo & 63;
          int bidx = m >> 11, n = m & 2047;
          size_t rowmajor = ((size_t)(bidx * NHEAD + head) * NN + n) * HDIM + hd;
          if (sec == 0)      qh[rowmajor] = (half_t)(c * 0.125f);  // fold 1/sqrt(64)
          else if (sec == 1) kh[rowmajor] = (half_t)c;
          else vt[((size_t)(bidx * NHEAD + head) * HDIM + hd) * NN + n] = (half_t)c;
        } else if (EPI == 1 || EPI == 3) {
          size_t idx = (size_t)m * N + o;
          outf[idx] = resid[idx] + c + bias[o];
        } else { // EPI == 2 : SiLU
          float tpre = c + bias[o];
          float sig = 1.0f / (1.0f + __expf(-tpre));
          outh[(size_t)m * N + o] = (half_t)(tpre * sig);
        }
      }
    }
  }
}

// ---------------------------------------------------------------------------
// Flash attention: one wave per (b, h, 16-row block). Full N=2048 columns,
// processed 32 at a time; online softmax in registers; P re-laid-out via LDS.
__global__ void __launch_bounds__(32) attn_kernel(const half_t* __restrict__ qh,
                                                  const half_t* __restrict__ kh,
                                                  const half_t* __restrict__ vt,
                                                  half_t* __restrict__ attn_h) {
  __shared__ _Float16 plds[16 * 32];
  const int lane = threadIdx.x;
  const int idx = blockIdx.x;
  const int i0 = (idx & 127) * 16;
  const int h  = (idx >> 7) & 15;
  const int b  = idx >> 11;

  const half_t* Q  = qh + (size_t)(b * NHEAD + h) * NN * HDIM;
  const half_t* Kp = kh + (size_t)(b * NHEAD + h) * NN * HDIM;
  const half_t* Vt = vt + (size_t)(b * NHEAD + h) * HDIM * NN;

  v16h a1 = load_a_tile(Q, HDIM, i0, 0, lane);
  v16h a2 = load_a_tile(Q, HDIM, i0, 32, lane);

  v8f zero = {};
  v8f oacc[4] = {zero, zero, zero, zero};
  float rmax[8], rsum[8];
#pragma unroll
  for (int vg = 0; vg < 8; ++vg) { rmax[vg] = -__builtin_inff(); rsum[vg] = 0.f; }

  for (int j0 = 0; j0 < NN; j0 += 32) {
    v8f s0 = zero, s1 = zero;
    s0 = wmma_f16(a1, load_b_tile(Kp, HDIM, j0, 0, lane), s0);
    s0 = wmma_f16(a2, load_b_tile(Kp, HDIM, j0, 32, lane), s0);
    s1 = wmma_f16(a1, load_b_tile(Kp, HDIM, j0 + 16, 0, lane), s1);
    s1 = wmma_f16(a2, load_b_tile(Kp, HDIM, j0 + 16, 32, lane), s1);

#pragma unroll
    for (int vg = 0; vg < 8; ++vg) {
      float v0 = s0[vg], v1 = s1[vg];
      float mx = fmaxf(v0, v1);
#pragma unroll
      for (int msk = 8; msk >= 1; msk >>= 1) mx = fmaxf(mx, __shfl_xor(mx, msk, 32));
      float mnew = fmaxf(rmax[vg], mx);
      float fac = __expf(rmax[vg] - mnew);
      float p0 = __expf(v0 - mnew);
      float p1 = __expf(v1 - mnew);
      float ts = p0 + p1;
#pragma unroll
      for (int msk = 8; msk >= 1; msk >>= 1) ts += __shfl_xor(ts, msk, 32);
      rsum[vg] = rsum[vg] * fac + ts;
      rmax[vg] = mnew;
#pragma unroll
      for (int t = 0; t < 4; ++t) oacc[t][vg] *= fac;
      int mrow = vg + 8 * (lane >> 4);
      int col = lane & 15;
      plds[mrow * 32 + col]      = (_Float16)p0;
      plds[mrow * 32 + 16 + col] = (_Float16)p1;
    }
    asm volatile("s_wait_dscnt 0" ::: "memory");  // LDS stores visible before reload

    union { v16h v; v8h h2[2]; } pu;
    {
      int row = lane & 15;
      int kb = (lane >> 4) << 3;
      pu.h2[0] = *(const v8h*)(plds + row * 32 + kb);
      pu.h2[1] = *(const v8h*)(plds + row * 32 + kb + 16);
    }
#pragma unroll
    for (int t = 0; t < 4; ++t)
      oacc[t] = wmma_f16(pu.v, load_b_tile(Vt, NN, t * 16, j0, lane), oacc[t]);
  }

#pragma unroll
  for (int vg = 0; vg < 8; ++vg) {
    float inv = 1.0f / rsum[vg];
    size_t base = ((size_t)(b * NN + i0 + vg + 8 * (lane >> 4))) * DIM + h * HDIM;
#pragma unroll
    for (int t = 0; t < 4; ++t)
      attn_h[base + t * 16 + (lane & 15)] = (half_t)(oacc[t][vg] * inv);
  }
}

// ---------------------------------------------------------------------------
extern "C" void kernel_launch(void* const* d_in, const int* in_sizes, int n_in,
                              void* d_out, int out_size, void* d_ws, size_t ws_size,
                              hipStream_t stream) {
  const float* x      = (const float*)d_in[0];
  const float* qkv_w  = (const float*)d_in[1];
  const float* out_w  = (const float*)d_in[2];
  const float* out_b  = (const float*)d_in[3];
  const float* ff_w1  = (const float*)d_in[4];
  const float* ff_b1  = (const float*)d_in[5];
  const float* ff_w2  = (const float*)d_in[6];
  const float* ff_b2  = (const float*)d_in[7];
  const float* ln1_w  = (const float*)d_in[8];
  const float* ln1_b  = (const float*)d_in[9];
  const float* ln2_w  = (const float*)d_in[10];
  const float* ln2_b  = (const float*)d_in[11];
  float* outp = (float*)d_out;

  char* wsb = (char*)d_ws;
  size_t off = 0;
  auto alloc = [&](size_t bytes) -> void* {
    void* p = wsb + off;
    off += (bytes + 255) & ~(size_t)255;
    return p;
  };
  half_t* wqkv_h = (half_t*)alloc((size_t)3 * DIM * DIM * 2);
  half_t* wout_h = (half_t*)alloc((size_t)DIM * DIM * 2);
  half_t* w1_h   = (half_t*)alloc((size_t)MLPD * DIM * 2);
  half_t* w2_h   = (half_t*)alloc((size_t)DIM * MLPD * 2);
  half_t* xln_h  = (half_t*)alloc((size_t)ROWS * DIM * 2);   // reused for LN1/LN2
  half_t* q_h    = (half_t*)alloc((size_t)ROWS * DIM * 2);
  half_t* k_h    = (half_t*)alloc((size_t)ROWS * DIM * 2);
  half_t* v_t    = (half_t*)alloc((size_t)ROWS * DIM * 2);
  half_t* attn_h = (half_t*)alloc((size_t)ROWS * DIM * 2);
  float*  x1     = (float*)alloc((size_t)ROWS * DIM * 4);
  half_t* h_h    = (half_t*)alloc((size_t)ROWS * MLPD * 2);

  auto cvt = [&](const float* src, half_t* dst, int n) {
    f32_to_f16_kernel<<<(n + 255) / 256, 256, 0, stream>>>(src, dst, n);
  };
  cvt(qkv_w, wqkv_h, 3 * DIM * DIM);
  cvt(out_w, wout_h, DIM * DIM);
  cvt(ff_w1, w1_h, MLPD * DIM);
  cvt(ff_w2, w2_h, DIM * MLPD);

  // LN1
  ln_f16_kernel<<<ROWS, 256, 0, stream>>>(x, ln1_w, ln1_b, xln_h);

  // QKV: [4096,1024] x [3072,1024]^T, scatter to q/k/v
  gemm_wmma_kernel<0><<<dim3(3 * DIM / 64, ROWS / 128), 128, 0, stream>>>(
      xln_h, wqkv_h, nullptr, nullptr, nullptr, nullptr, q_h, k_h, v_t, DIM, 3 * DIM);

  // attention: B*H*(N/16) waves
  attn_kernel<<<BB * NHEAD * (NN / 16), 32, 0, stream>>>(q_h, k_h, v_t, attn_h);

  // out-proj + residual
  gemm_wmma_kernel<1><<<dim3(DIM / 64, ROWS / 128), 128, 0, stream>>>(
      attn_h, wout_h, out_b, x, x1, nullptr, nullptr, nullptr, nullptr, DIM, DIM);

  // LN2
  ln_f16_kernel<<<ROWS, 256, 0, stream>>>(x1, ln2_w, ln2_b, xln_h);

  // MLP1 + SiLU
  gemm_wmma_kernel<2><<<dim3(MLPD / 64, ROWS / 128), 128, 0, stream>>>(
      xln_h, w1_h, ff_b1, nullptr, nullptr, h_h, nullptr, nullptr, nullptr, DIM, MLPD);

  // MLP2 + residual -> d_out
  gemm_wmma_kernel<3><<<dim3(DIM / 64, ROWS / 128), 128, 0, stream>>>(
      h_h, w2_h, ff_b2, x1, outp, nullptr, nullptr, nullptr, nullptr, MLPD, DIM);
}
